// MSTAGNN_82197084110974
// MI455X (gfx1250) — compile-verified, matrix operand order checked by reading, not run on hardware
//
#include <hip/hip_runtime.h>
#include <hip/hip_bf16.h>
#include <math.h>
#include <stdint.h>

#define NN    50000
#define EE    800000
#define FF    500
#define FP    512          // F padded to multiple of 32
#define HIDD  64
#define NHEAD 4
#define HC16  16
#define NC    8
#define NKHOP 2
#define CSTE  1e-5f
#define NPAD  50048        // 782 * 64 rows (GEMM tile padding)

typedef __attribute__((ext_vector_type(16))) __bf16 v16bf;
typedef __attribute__((ext_vector_type(8)))  __bf16 v8bf;
typedef __attribute__((ext_vector_type(8)))  float  v8f;

union ABfrag { v16bf v; v8bf h[2]; };

// ---- async global->LDS staging (CDNA5 ASYNCcnt path), with sync fallback ----
#if __has_builtin(__builtin_amdgcn_global_load_async_to_lds_b128)
typedef int v4i_vs __attribute__((vector_size(16)));
typedef __attribute__((address_space(1))) v4i_vs* gl_v4i_ptr;
typedef __attribute__((address_space(3))) v4i_vs* lds_v4i_ptr;
__device__ __forceinline__ void stage16(const void* g, void* s) {
  __builtin_amdgcn_global_load_async_to_lds_b128(
      (gl_v4i_ptr)(uintptr_t)g,
      (lds_v4i_ptr)(uint32_t)(uintptr_t)s,
      0, 0);
}
__device__ __forceinline__ void stage_wait() {
#if __has_builtin(__builtin_amdgcn_s_wait_asynccnt)
  __builtin_amdgcn_s_wait_asynccnt(0);
#else
  asm volatile("s_wait_asynccnt 0" ::: "memory");
#endif
}
#else
__device__ __forceinline__ void stage16(const void* g, void* s) {
  *(uint4*)s = *(const uint4*)g;
}
__device__ __forceinline__ void stage_wait() {}
#endif

// ---------------- prep: node_feat fp32 -> padded bf16 [NPAD, 512] ----------------
__global__ __launch_bounds__(256) void k_prepA(const float* __restrict__ nf,
                                               __bf16* __restrict__ Abf) {
  int tid = blockIdx.x * blockDim.x + threadIdx.x;   // NPAD*64 threads, 8 cols each
  if (tid >= NPAD * 64) return;
  int row = tid >> 6;
  int c0  = (tid & 63) << 3;
#pragma unroll
  for (int j = 0; j < 8; ++j) {
    int col = c0 + j;
    float v = (row < NN && col < FF) ? nf[(size_t)row * FF + col] : 0.f;
    Abf[(size_t)row * FP + col] = (__bf16)v;
  }
}

// ---- prep: pack Wi into per-lane B fragments  Bp[((kt*4+nt)*32+lane)*16+e] ----
__global__ __launch_bounds__(256) void k_prepB1(const float* __restrict__ Wi,
                                                __bf16* __restrict__ Bp) {
  int tid = blockIdx.x * blockDim.x + threadIdx.x;
  if (tid >= 16 * 4 * 32 * 16) return;
  int e    = tid & 15;
  int lane = (tid >> 4) & 31;
  int nt   = (tid >> 9) & 3;
  int kt   = tid >> 11;
  int n  = lane & 15;
  int kh = (lane >> 4) << 3;                 // 0 or 8
  int vg = e >> 1, w = e & 1;
  int kl = (vg < 4) ? (vg * 2 + w + kh) : (16 + (vg - 4) * 2 + w + kh);
  int k  = kt * 32 + kl;
  float v = (k < FF) ? Wi[k * HIDD + nt * 16 + n] : 0.f;
  Bp[tid] = (__bf16)v;
}

// ---- prep: pack Wq|Wk|Wv (160 output cols) into B fragments, K = 64 ----
__global__ __launch_bounds__(256) void k_prepB2(const float* __restrict__ Wq,
                                                const float* __restrict__ Wk,
                                                const float* __restrict__ Wv,
                                                __bf16* __restrict__ Bp) {
  int tid = blockIdx.x * blockDim.x + threadIdx.x;
  if (tid >= 2 * 10 * 32 * 16) return;
  int e    = tid & 15;
  int rest = tid >> 4;
  int lane = rest & 31;
  int r2   = rest >> 5;
  int nt   = r2 % 10;
  int kt   = r2 / 10;
  int n  = lane & 15;
  int kh = (lane >> 4) << 3;
  int vg = e >> 1, w = e & 1;
  int kl = (vg < 4) ? (vg * 2 + w + kh) : (16 + (vg - 4) * 2 + w + kh);
  int k  = kt * 32 + kl;                     // < 64 always
  int c  = nt * 16 + n;                      // 0..159
  float v;
  if (c < 64)       v = Wq[k * 64 + c];
  else if (c < 128) v = Wk[k * 64 + (c - 64)];
  else              v = Wv[k * 32 + (c - 128)];
  Bp[tid] = (__bf16)v;
}

// ------------- GEMM1: x = relu(A[NPAD,512] @ Wi + bi) -> bf16 [NPAD,64] -------------
// B (64KB) staged once into LDS (async) and shared by all 4 waves.
__global__ __launch_bounds__(128) void k_gemm1(const __bf16* __restrict__ Abf,
                                               const __bf16* __restrict__ Bp,
                                               const float* __restrict__ bi,
                                               __bf16* __restrict__ xbf) {
  __shared__ __align__(16) __bf16 Bs[16 * 4 * 32 * 16];   // 64 KB
  {
    const uint4* gsrc = (const uint4*)Bp;
    uint4* sdst = (uint4*)Bs;
#pragma unroll
    for (int i = 0; i < 32; ++i) {
      int idx = i * 128 + threadIdx.x;                    // 4096 x 16B
      stage16(gsrc + idx, sdst + idx);
    }
    stage_wait();
  }
  __syncthreads();

  const int lane    = threadIdx.x & 31;
  const int wave    = threadIdx.x >> 5;
  const int rowBase = blockIdx.x * 64 + wave * 16;
  const int laneM   = lane & 15;
  const int kh      = (lane >> 4) << 3;      // 0 or 8
  const __bf16* arow = Abf + (size_t)(rowBase + laneM) * FP + kh;
  v8f acc0 = {}, acc1 = {}, acc2 = {}, acc3 = {};
#pragma unroll 4
  for (int kt = 0; kt < 16; ++kt) {
    ABfrag a;
    a.h[0] = *(const v8bf*)(arow + kt * 32);
    a.h[1] = *(const v8bf*)(arow + kt * 32 + 16);
    __builtin_prefetch(arow + kt * 32 + 64, 0, 1);   // global_prefetch_b8: next A tile
    const __bf16* bp = Bs + kt * 2048 + lane * 16;
    v16bf b0 = *(const v16bf*)(bp);
    v16bf b1 = *(const v16bf*)(bp + 512);
    v16bf b2 = *(const v16bf*)(bp + 1024);
    v16bf b3 = *(const v16bf*)(bp + 1536);
    acc0 = __builtin_amdgcn_wmma_f32_16x16x32_bf16(false, a.v, false, b0, (short)0, acc0, false, false);
    acc1 = __builtin_amdgcn_wmma_f32_16x16x32_bf16(false, a.v, false, b1, (short)0, acc1, false, false);
    acc2 = __builtin_amdgcn_wmma_f32_16x16x32_bf16(false, a.v, false, b2, (short)0, acc2, false, false);
    acc3 = __builtin_amdgcn_wmma_f32_16x16x32_bf16(false, a.v, false, b3, (short)0, acc3, false, false);
  }
  v8f accs[4] = {acc0, acc1, acc2, acc3};
#pragma unroll
  for (int nt = 0; nt < 4; ++nt) {
    int col = nt * 16 + laneM;
    float bb = bi[col];
#pragma unroll
    for (int v = 0; v < 8; ++v) {
      float r = accs[nt][v] + bb;
      r = r > 0.f ? r : 0.f;
      xbf[(size_t)(rowBase + kh + v) * HIDD + col] = (__bf16)r;
    }
  }
}

// ------- GEMM2: [Q|K|V] = x[NPAD,64] @ [Wq|Wk|Wv]  (+1+elu on Q,K) -------
__global__ __launch_bounds__(128) void k_gemmqkv(const __bf16* __restrict__ xbf,
                                                 const __bf16* __restrict__ Bp,
                                                 const float* __restrict__ bq,
                                                 const float* __restrict__ bk,
                                                 const float* __restrict__ bvv,
                                                 float* __restrict__ Q,
                                                 float* __restrict__ Kf,
                                                 float* __restrict__ V) {
  __shared__ __align__(16) __bf16 Bs[2 * 10 * 32 * 16];   // 20 KB
  {
    const uint4* gsrc = (const uint4*)Bp;
    uint4* sdst = (uint4*)Bs;
#pragma unroll
    for (int i = 0; i < 10; ++i) {
      int idx = i * 128 + threadIdx.x;                    // 1280 x 16B
      stage16(gsrc + idx, sdst + idx);
    }
    stage_wait();
  }
  __syncthreads();

  const int lane    = threadIdx.x & 31;
  const int wave    = threadIdx.x >> 5;
  const int rowBase = blockIdx.x * 64 + wave * 16;
  const int laneM   = lane & 15;
  const int kh      = (lane >> 4) << 3;
  const __bf16* arow = xbf + (size_t)(rowBase + laneM) * HIDD + kh;
  v8f acc[10] = {};
#pragma unroll
  for (int kt = 0; kt < 2; ++kt) {
    ABfrag a;
    a.h[0] = *(const v8bf*)(arow + kt * 32);
    a.h[1] = *(const v8bf*)(arow + kt * 32 + 16);
#pragma unroll
    for (int nt = 0; nt < 10; ++nt) {
      v16bf b = *(const v16bf*)(Bs + ((kt * 10 + nt) * 32 + lane) * 16);
      acc[nt] = __builtin_amdgcn_wmma_f32_16x16x32_bf16(false, a.v, false, b, (short)0, acc[nt], false, false);
    }
  }
#pragma unroll
  for (int nt = 0; nt < 10; ++nt) {
    int c = nt * 16 + laneM;
#pragma unroll
    for (int v = 0; v < 8; ++v) {
      int row = rowBase + kh + v;
      float r = acc[nt][v];
      if (c < 64) {
        r += bq[c];
        r = r > 0.f ? 1.f + r : __expf(r);            // 1 + elu(z)
        Q[(size_t)row * 64 + c] = r;
      } else if (c < 128) {
        int cc = c - 64;
        r += bk[cc];
        r = r > 0.f ? 1.f + r : __expf(r);
        Kf[(size_t)row * 64 + cc] = r;
      } else {
        int cc = c - 128;
        V[(size_t)row * 32 + cc] = r + bvv[cc];
      }
    }
  }
}

// ---------------- degree + inverse ----------------
__global__ __launch_bounds__(256) void k_deg(const int* __restrict__ col,
                                             float* __restrict__ deg) {
  int e = blockIdx.x * blockDim.x + threadIdx.x;
  if (e < EE) atomicAdd(&deg[col[e]], 1.0f);
}
__global__ __launch_bounds__(256) void k_deginv(float* __restrict__ deg) {
  int n = blockIdx.x * blockDim.x + threadIdx.x;
  if (n < NN) { float d = deg[n]; deg[n] = d > 0.f ? 1.f / d : 0.f; }
}

// ---------------- M = outer(Kf, V) per node/head; hidden init ----------------
__global__ __launch_bounds__(256) void k_outer(const float* __restrict__ Kf,
                                               const float* __restrict__ V,
                                               float* __restrict__ M) {
  int tid = blockIdx.x * blockDim.x + threadIdx.x;   // NN*512
  if (tid >= NN * 512) return;
  int n = tid >> 9, u = tid & 511;
  int h = u >> 7, i = (u >> 3) & 15, j = u & 7;
  M[tid] = Kf[(size_t)n * 64 + h * 16 + i] * V[(size_t)n * 32 + h * 8 + j];
}
__global__ __launch_bounds__(256) void k_hidinit(const float* __restrict__ V,
                                                 const float* __restrict__ hopwise,
                                                 float* __restrict__ hid) {
  int tid = blockIdx.x * blockDim.x + threadIdx.x;   // NN*32
  if (tid < NN * 32) hid[tid] = V[tid] * hopwise[0];
}

// ---------------- teleport sums: tM = sum_n outer(Kf,V), tK = sum_n Kf ----------------
__global__ __launch_bounds__(256) void k_telsum(const float* __restrict__ Kf,
                                                const float* __restrict__ V,
                                                float* __restrict__ tM,
                                                float* __restrict__ tK) {
  int t  = threadIdx.x;
  int n0 = blockIdx.x * 64;
  int n1 = n0 + 64; if (n1 > NN) n1 = NN;
  int u1 = t, u2 = t + 256;
  int h1 = u1 >> 7, i1 = (u1 >> 3) & 15, j1 = u1 & 7;
  int h2 = u2 >> 7, i2 = (u2 >> 3) & 15, j2 = u2 & 7;
  float a1 = 0.f, a2 = 0.f, a3 = 0.f;
  for (int n = n0; n < n1; ++n) {
    const float* kr = Kf + (size_t)n * 64;
    const float* vr = V + (size_t)n * 32;
    a1 += kr[h1 * 16 + i1] * vr[h1 * 8 + j1];
    a2 += kr[h2 * 16 + i2] * vr[h2 * 8 + j2];
    if (t < 64) a3 += kr[t];
  }
  atomicAdd(&tM[u1], a1);
  atomicAdd(&tM[u2], a2);
  if (t < 64) atomicAdd(&tK[t], a3);
}

// ---------------- teleport attention term per node ----------------
__global__ __launch_bounds__(256) void k_telH(const float* __restrict__ Q,
                                              const float* __restrict__ tM,
                                              const float* __restrict__ tK,
                                              float* __restrict__ telH) {
  int n = blockIdx.x * blockDim.x + threadIdx.x;
  if (n >= NN) return;
  const float invN = 1.0f / (float)NN;
  float out[8] = {0.f, 0.f, 0.f, 0.f, 0.f, 0.f, 0.f, 0.f};
#pragma unroll
  for (int h = 0; h < NHEAD; ++h) {
    const float* q = Q + (size_t)n * 64 + h * 16;
    float den = 0.f;
#pragma unroll
    for (int i = 0; i < 16; ++i) den += q[i] * tK[h * 16 + i];
    den = den * invN + CSTE;
    float num[8] = {0.f, 0.f, 0.f, 0.f, 0.f, 0.f, 0.f, 0.f};
#pragma unroll
    for (int i = 0; i < 16; ++i) {
      float qi = q[i];
      const float* m = tM + (h * 16 + i) * 8;
#pragma unroll
      for (int j = 0; j < 8; ++j) num[j] += qi * m[j];
    }
#pragma unroll
    for (int j = 0; j < 8; ++j) out[j] += num[j] * invN / den;
  }
#pragma unroll
  for (int j = 0; j < 8; ++j) telH[(size_t)n * 8 + j] = out[j];
}

// ---------------- edge scatter: dst[col] += src[row] * deg_inv[row] ----------------
__global__ __launch_bounds__(128) void k_scatter(const int* __restrict__ row,
                                                 const int* __restrict__ col,
                                                 const float* __restrict__ dinv,
                                                 const float* __restrict__ Msrc,
                                                 const float* __restrict__ Ksrc,
                                                 float* __restrict__ Mdst,
                                                 float* __restrict__ Kdst) {
  int e = blockIdx.x;
  int t = threadIdx.x;
  int r = row[e], c = col[e];
  float w = dinv[r];
  float4 m = ((const float4*)(Msrc + (size_t)r * 512))[t];
  float* md = Mdst + (size_t)c * 512 + t * 4;
  atomicAdd(md + 0, m.x * w);
  atomicAdd(md + 1, m.y * w);
  atomicAdd(md + 2, m.z * w);
  atomicAdd(md + 3, m.w * w);
  if (t < 16) {
    float4 kk = ((const float4*)(Ksrc + (size_t)r * 64))[t];
    float* kd = Kdst + (size_t)c * 64 + t * 4;
    atomicAdd(kd + 0, kk.x * w);
    atomicAdd(kd + 1, kk.y * w);
    atomicAdd(kd + 2, kk.z * w);
    atomicAdd(kd + 3, kk.w * w);
  }
}

// ---------------- per-hop combine: hidden += gamma * (Q.M)/(Q.Kf + CST) ----------------
__global__ __launch_bounds__(256) void k_combine(const float* __restrict__ Q,
                                                 const float* __restrict__ M,
                                                 const float* __restrict__ Kf,
                                                 const float* __restrict__ hopwise,
                                                 const float* __restrict__ headwise,
                                                 int hop,
                                                 float* __restrict__ hid) {
  int tid = blockIdx.x * blockDim.x + threadIdx.x;   // NN*32
  if (tid >= NN * 32) return;
  int n = tid >> 5, hj = tid & 31, h = hj >> 3, j = hj & 7;
  const float* q = Q + (size_t)n * 64 + h * 16;
  const float* m = M + (size_t)n * 512 + h * 128 + j;
  const float* k = Kf + (size_t)n * 64 + h * 16;
  float num = 0.f, den = 0.f;
#pragma unroll
  for (int i = 0; i < 16; ++i) {
    num += q[i] * m[i * 8];
    den += q[i] * k[i];
  }
  den += CSTE;
  float e0 = __expf(headwise[0 * NKHOP + hop]);
  float e1 = __expf(headwise[1 * NKHOP + hop]);
  float e2 = __expf(headwise[2 * NKHOP + hop]);
  float e3 = __expf(headwise[3 * NKHOP + hop]);
  float es = e0 + e1 + e2 + e3;
  float eh = (h == 0) ? e0 : (h == 1) ? e1 : (h == 2) ? e2 : e3;
  float gamma = hopwise[hop + 1] * (eh / es);
  hid[tid] += gamma * (num / den);
}

// ---------------- final projection + teleport ----------------
__global__ __launch_bounds__(256) void k_final(const float* __restrict__ hid,
                                               const float* __restrict__ Wo,
                                               const float* __restrict__ bo,
                                               const float* __restrict__ telH,
                                               const float* __restrict__ teleport,
                                               float* __restrict__ out) {
  int tid = blockIdx.x * blockDim.x + threadIdx.x;   // NN*8
  if (tid >= NN * 8) return;
  int n = tid >> 3, c = tid & 7;
  const float* hrow = hid + (size_t)n * 32;
  float s = bo[c];
#pragma unroll
  for (int k = 0; k < 32; ++k) s += hrow[k] * Wo[k * 8 + c];
  out[tid] = s + teleport[0] * telH[tid];
}

extern "C" void kernel_launch(void* const* d_in, const int* in_sizes, int n_in,
                              void* d_out, int out_size, void* d_ws, size_t ws_size,
                              hipStream_t stream) {
  const float* node_feat = (const float*)d_in[0];
  const int*   eidx      = (const int*)d_in[1];
  const float* Wi = (const float*)d_in[2];
  const float* bi = (const float*)d_in[3];
  const float* Wq = (const float*)d_in[4];
  const float* bq = (const float*)d_in[5];
  const float* Wk = (const float*)d_in[6];
  const float* bk = (const float*)d_in[7];
  const float* Wv = (const float*)d_in[8];
  const float* bv = (const float*)d_in[9];
  const float* Wo = (const float*)d_in[10];
  const float* bo = (const float*)d_in[11];
  const float* hopwise  = (const float*)d_in[12];
  const float* headwise = (const float*)d_in[13];
  const float* teleport = (const float*)d_in[14];
  const int* rowI = eidx;
  const int* colI = eidx + EE;
  float* out = (float*)d_out;

  char* p = (char*)d_ws;
  auto carve = [&](size_t bytes) -> char* {
    char* r = p;
    p += (bytes + 255) & ~(size_t)255;
    return r;
  };
  __bf16* Abf  = (__bf16*)carve((size_t)NPAD * FP * 2);
  __bf16* xbf  = (__bf16*)carve((size_t)NPAD * HIDD * 2);
  __bf16* Bp1  = (__bf16*)carve((size_t)16 * 4 * 32 * 16 * 2);
  __bf16* Bp2  = (__bf16*)carve((size_t)2 * 10 * 32 * 16 * 2);
  float*  Q    = (float*)carve((size_t)NPAD * 64 * 4);
  float*  Kf0  = (float*)carve((size_t)NPAD * 64 * 4);
  float*  V    = (float*)carve((size_t)NPAD * 32 * 4);
  float*  M0   = (float*)carve((size_t)NN * 512 * 4);
  float*  M1   = (float*)carve((size_t)NN * 512 * 4);
  float*  Kf1  = (float*)carve((size_t)NN * 64 * 4);
  float*  hid  = (float*)carve((size_t)NN * 32 * 4);
  float*  deg  = (float*)carve((size_t)NN * 4);
  float*  tM   = (float*)carve(512 * 4);
  float*  tK   = (float*)carve(64 * 4);
  float*  telH = (float*)carve((size_t)NN * 8 * 4);

  const int B256 = 256;

  // --- prep: bf16 packing + degree ---
  (void)hipMemsetAsync(deg, 0, (size_t)NN * 4, stream);
  k_prepA<<<(NPAD * 64 + B256 - 1) / B256, B256, 0, stream>>>(node_feat, Abf);
  k_prepB1<<<(16 * 4 * 32 * 16 + B256 - 1) / B256, B256, 0, stream>>>(Wi, Bp1);
  k_prepB2<<<(2 * 10 * 32 * 16 + B256 - 1) / B256, B256, 0, stream>>>(Wq, Wk, Wv, Bp2);
  k_deg<<<(EE + B256 - 1) / B256, B256, 0, stream>>>(colI, deg);
  k_deginv<<<(NN + B256 - 1) / B256, B256, 0, stream>>>(deg);

  // --- WMMA GEMMs ---
  k_gemm1<<<NPAD / 64, 128, 0, stream>>>(Abf, Bp1, bi, xbf);
  k_gemmqkv<<<NPAD / 64, 128, 0, stream>>>(xbf, Bp2, bq, bk, bv, Q, Kf0, V);

  // --- M = outer(Kf,V); hidden = V*hopwise0; teleport term ---
  k_outer<<<(NN * 512 + B256 - 1) / B256, B256, 0, stream>>>(Kf0, V, M0);
  k_hidinit<<<(NN * 32 + B256 - 1) / B256, B256, 0, stream>>>(V, hopwise, hid);
  (void)hipMemsetAsync(tM, 0, 512 * 4, stream);
  (void)hipMemsetAsync(tK, 0, 64 * 4, stream);
  k_telsum<<<(NN + 63) / 64, B256, 0, stream>>>(Kf0, V, tM, tK);
  k_telH<<<(NN + B256 - 1) / B256, B256, 0, stream>>>(Q, tM, tK, telH);

  // --- hop 1: M0,Kf0 -> M1,Kf1 ---
  (void)hipMemsetAsync(M1, 0, (size_t)NN * 512 * 4, stream);
  (void)hipMemsetAsync(Kf1, 0, (size_t)NN * 64 * 4, stream);
  k_scatter<<<EE, 128, 0, stream>>>(rowI, colI, deg, M0, Kf0, M1, Kf1);
  k_combine<<<(NN * 32 + B256 - 1) / B256, B256, 0, stream>>>(Q, M1, Kf1, hopwise, headwise, 0, hid);

  // --- hop 2: M1,Kf1 -> M0,Kf0 (teleport already consumed M0/Kf0) ---
  (void)hipMemsetAsync(M0, 0, (size_t)NN * 512 * 4, stream);
  (void)hipMemsetAsync(Kf0, 0, (size_t)NN * 64 * 4, stream);
  k_scatter<<<EE, 128, 0, stream>>>(rowI, colI, deg, M1, Kf1, M0, Kf0);
  k_combine<<<(NN * 32 + B256 - 1) / B256, B256, 0, stream>>>(Q, M0, Kf0, hopwise, headwise, 1, hid);

  // --- output projection + teleport ---
  k_final<<<(NN * 8 + B256 - 1) / B256, B256, 0, stream>>>(hid, Wo, bo, telH, teleport, out);
}
